// FlowWaterGVP_66314295050607
// MI455X (gfx1250) — matrix operand order repaired and checked by reading, not compiled, not run on hardware
//
#include <hip/hip_runtime.h>
#include <math.h>

// ---------------------------------------------------------------------------
// FlowWaterGVP for MI455X (gfx1250, wave32).
// Dense GEMMs run on v_wmma_f32_16x16x32_bf16 (f32 accumulate). All A-side
// activation buffers are built with K padded to a multiple of 32 and rows
// 16B-aligned, so the A tile is staged into LDS with
// global_load_async_to_lds_b128 (ASYNCcnt) -- the CDNA5 async global->LDS
// path -- and WMMA fragments are exactly two ds_load_b128 per matrix.
// ---------------------------------------------------------------------------

typedef __bf16 bf16;
typedef __attribute__((ext_vector_type(16))) __bf16 v16bf;
typedef __attribute__((ext_vector_type(8)))  __bf16 v8bf;
typedef __attribute__((ext_vector_type(8)))  float  v8f;

#define GS_LOOP(i, n) for (long i = (long)blockIdx.x * blockDim.x + threadIdx.x; \
                           i < (n); i += (long)gridDim.x * blockDim.x)

// ------------------------------ elementwise --------------------------------

__global__ void k_f32_to_bf16(const float* __restrict__ in, bf16* __restrict__ out, long n) {
  GS_LOOP(i, n) out[i] = (bf16)in[i];
}

// B-weight conversion with zero row padding: dst is Kp x N, src K x N.
__global__ void k_pad_weight(const float* __restrict__ src, bf16* __restrict__ dst,
                             long K, int N, long Kp) {
  GS_LOOP(i, Kp * (long)N) {
    long r = i / N; int c = (int)(i % N);
    dst[i] = (r < K) ? (bf16)src[r * (long)N + c] : (bf16)0.f;
  }
}

__global__ void k_copy_f32_bf16_mat(const float* __restrict__ src, int sld,
                                    bf16* __restrict__ dst, int dld, int doff,
                                    long rows, int cols) {
  GS_LOOP(i, rows * (long)cols) {
    long r = i / cols; int c = (int)(i % cols);
    dst[r * (long)dld + doff + c] = (bf16)src[r * (long)sld + c];
  }
}

__global__ void k_zero_cols_bf16(bf16* __restrict__ dst, int ld, int from, int to, long rows) {
  int w = to - from;
  GS_LOOP(i, rows * (long)w) {
    long r = i / w; int c = (int)(i % w);
    dst[r * (long)ld + from + c] = (bf16)0.f;
  }
}

__global__ void k_relu(float* __restrict__ x, long n) {
  GS_LOOP(i, n) x[i] = fmaxf(x[i], 0.f);
}

__global__ void k_add(float* __restrict__ dst, const float* __restrict__ src, long n) {
  GS_LOOP(i, n) dst[i] += src[i];
}

__global__ void k_sigmoid_bf16(const float* __restrict__ in, bf16* __restrict__ out, long n) {
  GS_LOOP(i, n) out[i] = (bf16)(1.f / (1.f + expf(-in[i])));
}

// --------------------------------- GEMM ------------------------------------
// C(MxN,f32) = A(MxK,bf16) * B(KxN,bf16) [+bias] [+act]; act: 0 none,1 relu,2 gelu
// REQUIRES: K % 32 == 0, lda == K (packed padded rows), B has K (padded) rows.

__global__ __launch_bounds__(256)
void k_gemm(const bf16* __restrict__ A, int lda,
            const bf16* __restrict__ B, int ldb,
            const float* __restrict__ bias,
            float* __restrict__ C, int ldc,
            int M, int N, int K, int act) {
  __shared__ __align__(16) bf16 As[128 * 32];   // row-major MxK tile
  __shared__ __align__(16) bf16 Bst[32 * 32];   // TRANSPOSED: Bst[n][k]
  const int tid  = threadIdx.x;
  const int lane = tid & 31, wave = tid >> 5;
  const int wm = wave >> 1, wn = wave & 1;      // 4x2 wave grid; wave = 32x16 out
  const int bm = blockIdx.x * 128, bn = blockIdx.y * 32;
  const int m  = lane & 15, hi = lane >> 4;
  v8f acc0 = {0.f, 0.f, 0.f, 0.f, 0.f, 0.f, 0.f, 0.f};
  v8f acc1 = {0.f, 0.f, 0.f, 0.f, 0.f, 0.f, 0.f, 0.f};

  for (int k0 = 0; k0 < K; k0 += 32) {
    // A tile: 128x32 bf16 = 512 aligned 16B chunks via async global->LDS DMA.
    for (int i = tid; i < 512; i += 256) {
      int r = i >> 2, cc = i & 3;
      int gr = bm + r;
      bf16* lp = As + i * 8;
      if (gr < M) {
        const bf16* gp = A + (long)gr * lda + k0 + cc * 8;
        unsigned lo = (unsigned)(unsigned long long)lp;
        asm volatile("global_load_async_to_lds_b128 %0, %1, off"
                     :: "v"(lo), "v"((unsigned long long)gp)
                     : "memory");
      } else {
        v8bf z = {};
        *(v8bf*)lp = z;
      }
    }
    // B tile (transpose while staging; B rows are K-padded so no K guard).
    for (int i = tid; i < 32 * 32; i += 256) {
      int r = i >> 5, c = i & 31;           // r = K-row, c = N-col of tile
      int gc = bn + c;
      Bst[c * 32 + r] = (gc < N) ? B[(long)(k0 + r) * ldb + gc] : (bf16)0.f;
    }
    asm volatile("s_wait_asynccnt 0x0" ::: "memory");
    __syncthreads();

    // Fragments per CDNA5 ISA 7.12.2: A lane row = wm*32+t*16+m, elements are
    // two contiguous 8-wide runs (K = hi*8.., K = 16+hi*8..). B (transposed
    // staging) uses the same two 8-wide runs (K = hi*8.., K = 16+hi*8..) of column wn*16+m.
    const bf16* bp = Bst + (wn * 16 + m) * 32 + hi * 8;
    v8bf b0 = *(const v8bf*)bp;
    v8bf b1 = *(const v8bf*)(bp + 16);
    v16bf bfrag = __builtin_shufflevector(b0, b1, 0, 1, 2, 3, 4, 5, 6, 7,
                                          8, 9, 10, 11, 12, 13, 14, 15);
    const bf16* ap0 = As + (wm * 32 + m) * 32 + hi * 8;
    v8bf a00 = *(const v8bf*)ap0;
    v8bf a01 = *(const v8bf*)(ap0 + 16);
    v16bf af0 = __builtin_shufflevector(a00, a01, 0, 1, 2, 3, 4, 5, 6, 7,
                                        8, 9, 10, 11, 12, 13, 14, 15);
    acc0 = __builtin_amdgcn_wmma_f32_16x16x32_bf16(false, af0, false, bfrag,
                                                   (short)0, acc0, false, false);
    const bf16* ap1 = ap0 + 16 * 32;
    v8bf a10 = *(const v8bf*)ap1;
    v8bf a11 = *(const v8bf*)(ap1 + 16);
    v16bf af1 = __builtin_shufflevector(a10, a11, 0, 1, 2, 3, 4, 5, 6, 7,
                                        8, 9, 10, 11, 12, 13, 14, 15);
    acc1 = __builtin_amdgcn_wmma_f32_16x16x32_bf16(false, af1, false, bfrag,
                                                   (short)0, acc1, false, false);
    __syncthreads();
  }

  const int col = bn + wn * 16 + m;
  if (col < N) {
    float bv = bias ? bias[col] : 0.f;
#pragma unroll
    for (int t = 0; t < 2; ++t) {
      v8f acc = t ? acc1 : acc0;
#pragma unroll
      for (int r = 0; r < 8; ++r) {
        int row = bm + wm * 32 + t * 16 + (hi << 3) + r;  // C/D VGPR layout
        if (row < M) {
          float x = acc[r] + bv;
          if (act == 1) x = fmaxf(x, 0.f);
          else if (act == 2) x = 0.5f * x * (1.f + erff(x * 0.70710678f));
          C[(long)row * ldc + col] = x;
        }
      }
    }
  }
}

// --------------------------------- KNN -------------------------------------

template <int K>
__global__ __launch_bounds__(128)
void k_knn(const float* __restrict__ q, const float* __restrict__ kp,
           int nq, int nk, int excl, int* __restrict__ nbr) {
  int qi = blockIdx.x; if (qi >= nq) return;
  float qx = q[3 * qi], qy = q[3 * qi + 1], qz = q[3 * qi + 2];
  float bd[K]; int bi[K];
#pragma unroll
  for (int t = 0; t < K; ++t) { bd[t] = 3.4e38f; bi[t] = 0; }
  for (int j = threadIdx.x; j < nk; j += 128) {
    if (excl && j == qi) continue;
    float dx = kp[3 * j] - qx, dy = kp[3 * j + 1] - qy, dz = kp[3 * j + 2] - qz;
    float d2 = dx * dx + dy * dy + dz * dz;
    if (d2 < bd[K - 1]) {
      int pos = K - 1;
#pragma unroll
      for (int t = K - 2; t >= 0; --t) if (d2 < bd[t]) pos = t;
#pragma unroll
      for (int t = K - 1; t > 0; --t) if (t > pos) { bd[t] = bd[t - 1]; bi[t] = bi[t - 1]; }
#pragma unroll
      for (int t = 0; t < K; ++t) if (t == pos) { bd[t] = d2; bi[t] = j; }
    }
  }
  __shared__ float sd[128 * K];
  __shared__ int   sidr[128 * K];
  __shared__ int   heads[128];
  int tt = threadIdx.x;
#pragma unroll
  for (int t = 0; t < K; ++t) { sd[tt * K + t] = bd[t]; sidr[tt * K + t] = bi[t]; }
  heads[tt] = 0;
  __syncthreads();
  if (tt == 0) {
    for (int o = 0; o < K; ++o) {
      float best = 3.5e38f; int bl = 0;
      for (int l = 0; l < 128; ++l) {
        int h = heads[l];
        if (h < K && sd[l * K + h] < best) { best = sd[l * K + h]; bl = l; }
      }
      nbr[(long)qi * K + o] = sidr[bl * K + heads[bl]];
      heads[bl]++;
    }
  }
}

__global__ void k_make_srcidx(int* __restrict__ out, long E, int K) {
  GS_LOOP(i, E) out[i] = (int)(i / K);
}

// ----------------------------- edge geometry -------------------------------

__global__ void k_edge_geom(const float* __restrict__ pos_s, const float* __restrict__ pos_d,
                            const int* __restrict__ sidx, const int* __restrict__ didx,
                            long E, int R, float* __restrict__ rbf, float* __restrict__ unit) {
  GS_LOOP(e, E) {
    int s = sidx[e], d = didx[e];
    float dx = pos_s[3 * s] - pos_d[3 * d];
    float dy = pos_s[3 * s + 1] - pos_d[3 * d + 1];
    float dz = pos_s[3 * s + 2] - pos_d[3 * d + 2];
    float dist = fmaxf(sqrtf(dx * dx + dy * dy + dz * dz), 1e-8f);
    unit[3 * e] = dx / dist; unit[3 * e + 1] = dy / dist; unit[3 * e + 2] = dz / dist;
    const float c = 20.f;
    float xs = fmaxf(dist, 1e-8f);
    float mask = (xs / c < 1.f) ? 1.f : 0.f;
    float pre = sqrtf(2.f / c) / xs * mask;
    for (int r = 0; r < R; ++r)
      rbf[e * R + r] = pre * sinf((float)(r + 1) * 3.14159265358979f * xs / c);
  }
}

// --------------------------- gather / concat -------------------------------
// Writes [0, si) data and zeros the pad [si+h, ld); [si, si+h) (vn) is
// written later by k_vh_post.

__global__ void k_build_msg_s(const float* __restrict__ s_src, const float* __restrict__ s_dst, int S,
                              const float* __restrict__ rbf, int R,
                              const int* __restrict__ sidx, const int* __restrict__ didx,
                              long E, int si, int h, int ld, bf16* __restrict__ dst) {
  GS_LOOP(i, E * (long)ld) {
    long e = i / ld; int c = (int)(i % ld);
    if (c >= si && c < si + h) continue;   // vn region
    float v = 0.f;
    if (c < S)          v = s_src[(long)sidx[e] * S + c];
    else if (c < S + R) v = rbf[e * R + (c - S)];
    else if (c < si)    v = s_dst[(long)didx[e] * S + (c - S - R)];
    dst[e * (long)ld + c] = (bf16)v;
  }
}

// v_in (coord-major 3E x vip, bf16): [v_src | unit | v_dst | 0-pad]
__global__ void k_build_msg_v(const float* __restrict__ v_src, const float* __restrict__ v_dst, int V,
                              const float* __restrict__ unit,
                              const int* __restrict__ sidx, const int* __restrict__ didx,
                              long E, int vip, bf16* __restrict__ dst) {
  int cols = 2 * V + 1;
  GS_LOOP(i, 3 * E * (long)vip) {
    long r = i / vip; int j = (int)(i % vip);
    int c = (int)(r / E); long e = r % E;
    float v = 0.f;
    if (j < V)          v = v_src[((long)sidx[e] * V + j) * 3 + c];
    else if (j == V)    v = unit[e * 3 + c];
    else if (j < cols)  v = v_dst[((long)didx[e] * V + (j - V - 1)) * 3 + c];
    dst[i] = (bf16)v;
  }
}

__global__ void k_v_to_cm(const float* __restrict__ v, long n, int V, int vip,
                          bf16* __restrict__ dst) {
  GS_LOOP(i, 3 * n * (long)vip) {
    long r = i / vip; int j = (int)(i % vip);
    int c = (int)(r / n); long e = r % n;
    dst[i] = (j < V) ? (bf16)v[((long)e * V + j) * 3 + c] : (bf16)0.f;
  }
}

__global__ void k_concat_v_cm(const float* __restrict__ v1, int V1,
                              const float* __restrict__ v2, int V2,
                              long n, int vip, bf16* __restrict__ dst) {
  GS_LOOP(i, 3 * n * (long)vip) {
    long r = i / vip; int j = (int)(i % vip);
    int c = (int)(r / n); long e = r % n;
    float x = 0.f;
    if (j < V1)            x = v1[((long)e * V1 + j) * 3 + c];
    else if (j < V1 + V2)  x = v2[((long)e * V2 + (j - V1)) * 3 + c];
    dst[i] = (bf16)x;
  }
}

__global__ void k_concat_col_bf16(const float* __restrict__ x, int F,
                                  const float* __restrict__ t, long n, int ldp,
                                  bf16* __restrict__ dst) {
  GS_LOOP(i, n * (long)ldp) {
    long e = i / ldp; int c = (int)(i % ldp);
    float v = 0.f;
    if (c < F)       v = x[e * F + c];
    else if (c == F) v = t[0];
    dst[i] = (bf16)v;
  }
}

// ------------------------------ GVP pieces ---------------------------------

// vn = ||Vh|| appended to s_in (bf16); also bf16 K-padded copy of Vh (3E x hp).
__global__ void k_vh_post(const float* __restrict__ Vh, long E, int h, int hp,
                          bf16* __restrict__ sdst, int ld, int off,
                          bf16* __restrict__ Vhb) {
  GS_LOOP(i, E * (long)hp) {
    long e = i / hp; int j = (int)(i % hp);
    if (j < h) {
      float a = Vh[((long)0 * E + e) * h + j];
      float b = Vh[((long)1 * E + e) * h + j];
      float c = Vh[((long)2 * E + e) * h + j];
      sdst[e * (long)ld + off + j] = (bf16)sqrtf(a * a + b * b + c * c + 1e-8f);
      Vhb[((long)0 * E + e) * hp + j] = (bf16)a;
      Vhb[((long)1 * E + e) * hp + j] = (bf16)b;
      Vhb[((long)2 * E + e) * hp + j] = (bf16)c;
    } else {
      Vhb[((long)0 * E + e) * hp + j] = (bf16)0.f;
      Vhb[((long)1 * E + e) * hp + j] = (bf16)0.f;
      Vhb[((long)2 * E + e) * hp + j] = (bf16)0.f;
    }
  }
}

// V_out[e,j,c] = Vu[c*E+e, j] * sigmoid(gate[e,j])   (node-major out)
__global__ void k_gate_apply(const float* __restrict__ Vu, const float* __restrict__ gate,
                             long E, int vo, float* __restrict__ Vout) {
  GS_LOOP(i, E * (long)vo) {
    long e = i / vo; int j = (int)(i % vo);
    float g = 1.f / (1.f + expf(-gate[e * vo + j]));
    for (int c = 0; c < 3; ++c)
      Vout[((long)e * vo + j) * 3 + c] = Vu[((long)c * E + e) * vo + j] * g;
  }
}

__global__ void k_scatter_add(const float* __restrict__ src, const int* __restrict__ didx,
                              long E, int F, float* __restrict__ out) {
  GS_LOOP(i, E * (long)F) {
    long e = i / F; int f = (int)(i % F);
    atomicAdd(&out[(long)didx[e] * F + f], src[i]);
  }
}

// ------------------------- layernorm / conditioning ------------------------

__global__ __launch_bounds__(256)
void k_layernorm256(const float* __restrict__ x, const float* __restrict__ g,
                    const float* __restrict__ b, float* __restrict__ out, int n) {
  __shared__ float red[256];
  int row = blockIdx.x; if (row >= n) return;
  int c = threadIdx.x;
  float v = x[(long)row * 256 + c];
  red[c] = v; __syncthreads();
  for (int s = 128; s > 0; s >>= 1) { if (c < s) red[c] += red[c + s]; __syncthreads(); }
  float mu = red[0] / 256.f; __syncthreads();
  float d = v - mu;
  red[c] = d * d; __syncthreads();
  for (int s = 128; s > 0; s >>= 1) { if (c < s) red[c] += red[c + s]; __syncthreads(); }
  float var = red[0] / 256.f;
  out[(long)row * 256 + c] = d * rsqrtf(var + 1e-5f) * g[c] + b[c];
}

__global__ __launch_bounds__(256)
void k_sc_scalar(const float* __restrict__ x1, const float* __restrict__ pw,
                 const float* __restrict__ W, const float* __restrict__ bb,
                 const float* __restrict__ lg, const float* __restrict__ lb,
                 int n, float* __restrict__ s_w) {
  __shared__ float red[256];
  int e = blockIdx.x; if (e >= n) return;
  int c = threadIdx.x;
  float dx = x1[3 * e] - pw[3 * e], dy = x1[3 * e + 1] - pw[3 * e + 1], dz = x1[3 * e + 2] - pw[3 * e + 2];
  float dm = sqrtf(dx * dx + dy * dy + dz * dz + 1e-12f);
  float v = dm * W[c] + bb[c];
  v = 0.5f * v * (1.f + erff(v * 0.70710678f));
  red[c] = v; __syncthreads();
  for (int s = 128; s > 0; s >>= 1) { if (c < s) red[c] += red[c + s]; __syncthreads(); }
  float mu = red[0] / 256.f; __syncthreads();
  float d = v - mu;
  red[c] = d * d; __syncthreads();
  for (int s = 128; s > 0; s >>= 1) { if (c < s) red[c] += red[c + s]; __syncthreads(); }
  float var = red[0] / 256.f;
  s_w[(long)e * 256 + c] += d * rsqrtf(var + 1e-5f) * lg[c] + lb[c];
}

// sc_vec GVP: v_w += delta * (Wh@Wv)[j] * sigmoid(bg[j])
__global__ __launch_bounds__(256)
void k_self_cond_vec(const float* __restrict__ x1, const float* __restrict__ pw,
                     const float* __restrict__ Wh, const float* __restrict__ Wv,
                     const float* __restrict__ bg, long n, float* __restrict__ v_w) {
  __shared__ float weff[32];
  if (threadIdx.x < 32) {
    float s = 0.f;
    for (int h = 0; h < 32; ++h) s += Wh[h] * Wv[h * 32 + threadIdx.x];
    weff[threadIdx.x] = s * (1.f / (1.f + expf(-bg[threadIdx.x])));
  }
  __syncthreads();
  GS_LOOP(i, n * 32) {
    long e = i / 32; int j = (int)(i % 32);
    float w = weff[j];
    v_w[((long)e * 32 + j) * 3 + 0] += (x1[3 * e] - pw[3 * e]) * w;
    v_w[((long)e * 32 + j) * 3 + 1] += (x1[3 * e + 1] - pw[3 * e + 1]) * w;
    v_w[((long)e * 32 + j) * 3 + 2] += (x1[3 * e + 2] - pw[3 * e + 2]) * w;
  }
}

// head GVP: out[e,c] = sigmoid(bg[0]) * sum_v v_w[e,v,c] * (Wh@Wv)[v]
__global__ __launch_bounds__(256)
void k_head(const float* __restrict__ v_w, const float* __restrict__ Wh,
            const float* __restrict__ Wv, const float* __restrict__ bg,
            long n, float* __restrict__ out) {
  __shared__ float weff[32];
  if (threadIdx.x < 32) {
    float s = 0.f;
    for (int h = 0; h < 32; ++h) s += Wh[threadIdx.x * 32 + h] * Wv[h];
    weff[threadIdx.x] = s;
  }
  __syncthreads();
  float gg = 1.f / (1.f + expf(-bg[0]));
  GS_LOOP(i, n * 3) {
    long e = i / 3; int c = (int)(i % 3);
    float s = 0.f;
    for (int v = 0; v < 32; ++v) s += v_w[((long)e * 32 + v) * 3 + c] * weff[v];
    out[i] = s * gg;
  }
}

// ------------------------------- host side ---------------------------------

namespace {
enum { cN_P = 5000, cN_W = 10000, cP_IN = 20, cW_IN = 16, cENC_S = 128, cENC_V = 16,
       cS_H = 256, cV_H = 32, cENC_RBF = 16, cEDGE_RBF = 32,
       cK_PP = 16, cK_PW = 12, cK_WW = 8 };
const long cE_PP = (long)cN_P * cK_PP;   // 80000
const long cE_PW = (long)cN_P * cK_PW;   // 60000
const long cE_WW = (long)cN_W * cK_WW;   // 80000

struct GVPp {
  const float *Wh, *Ws, *Wv, *Wg, *bg, *bs;
  const bf16 *Whb, *Wsb, *Wvb, *Wgb;
  int si, vi, so, vo, h;
  int ldw, vip, hp;   // padded K dims: ceil32(si+h), ceil32(vi), ceil32(h)
};

inline int pad32(int x) { return (x + 31) & ~31; }

inline dim3 ew(long n) {
  long b = (n + 255) / 256;
  if (b > 16384) b = 16384;
  if (b < 1) b = 1;
  return dim3((unsigned)b);
}
}  // namespace

extern "C" void kernel_launch(void* const* d_in, const int* in_sizes, int n_in,
                              void* d_out, int out_size, void* d_ws, size_t ws_size,
                              hipStream_t stream) {
  const float* protein_x = (const float*)d_in[0];
  const float* water_x   = (const float*)d_in[1];
  const float* pos_p     = (const float*)d_in[2];
  const float* pos_w     = (const float*)d_in[3];
  const float* t_in      = (const float*)d_in[4];
  const float* x1_pred   = (const float*)d_in[5];
  const int*   pp_ei     = (const int*)d_in[6];
  (void)in_sizes; (void)out_size; (void)ws_size;

  // Param walker: either one leaf per d_in slot (tree_leaves, sorted-key
  // recursive order), or a single flat blob at d_in[7].
  int pidx = 7;
  const float* flat = (n_in <= 8) ? (const float*)d_in[7] : nullptr;
  long foff = 0;
  auto nextParam = [&](long cnt) -> const float* {
    if (flat) { const float* p = flat + foff; foff += cnt; return p; }
    return (const float*)d_in[pidx++];
  };

  // Workspace arena.
  char* wsb = (char*)d_ws;
  size_t woff = 0;
  auto alloc = [&](size_t bytes) -> void* {
    size_t a = (woff + 255) & ~(size_t)255;
    woff = a + bytes;
    return (void*)(wsb + a);
  };
  auto tob16p = [&](const float* w, long K, int N, long Kp) -> const bf16* {
    bf16* d = (bf16*)alloc((size_t)Kp * N * sizeof(bf16));
    k_pad_weight<<<ew(Kp * N), 256, 0, stream>>>(w, d, K, N, Kp);
    return d;
  };
  auto fetch_gvp = [&](int si, int vi, int so, int vo) -> GVPp {
    GVPp g{}; g.si = si; g.vi = vi; g.so = so; g.vo = vo;
    g.h = vi ? (vi > vo ? vi : vo) : 0;
    g.ldw = pad32(si + g.h); g.vip = pad32(vi); g.hp = pad32(g.h);
    long sk = si + (vi ? g.h : 0);
    if (vo && so) g.Wg = nextParam((long)so * vo);       // sorted leaf order:
    if (vi)       g.Wh = nextParam((long)vi * g.h);      // Wg, Wh, Ws, Wv, bg, bs
    if (so)       g.Ws = nextParam(sk * so);
    if (vo)       g.Wv = nextParam((long)g.h * vo);
    if (vo)       g.bg = nextParam(vo);
    if (so)       g.bs = nextParam(so);
    if (g.Wg) g.Wgb = tob16p(g.Wg, so, vo, so);
    if (g.Wh) g.Whb = tob16p(g.Wh, vi, g.h, g.vip);
    if (g.Ws) g.Wsb = tob16p(g.Ws, sk, so, g.ldw);
    if (g.Wv) g.Wvb = tob16p(g.Wv, g.h, vo, g.hp);
    return g;
  };

  // ---- fetch params in sorted top-level key order ----
  // blocks < enc2flow < enc_blocks < enc_embed_W < enc_embed_b < head <
  // p_enc_W < p_enc_b < p_ln_b < p_ln_g < sc_ln_b < sc_ln_g < sc_sca_W <
  // sc_sca_b < sc_vec < w_enc_W < w_enc_b < w_ln_b < w_ln_g
  GVPp blocks[4][3][3];
  for (int b = 0; b < 4; ++b) {
    for (int L = 0; L < 3; ++L)  // pw_msg
      blocks[b][0][L] = L ? fetch_gvp(cS_H, cV_H, cS_H, cV_H)
                          : fetch_gvp(2 * cS_H + cEDGE_RBF, 2 * cV_H + 1, cS_H, cV_H);
    for (int L = 0; L < 3; ++L)  // upd
      blocks[b][1][L] = L ? fetch_gvp(cS_H, cV_H, cS_H, cV_H)
                          : fetch_gvp(2 * cS_H, 2 * cV_H, cS_H, cV_H);
    for (int L = 0; L < 3; ++L)  // ww_msg
      blocks[b][2][L] = L ? fetch_gvp(cS_H, cV_H, cS_H, cV_H)
                          : fetch_gvp(2 * cS_H + cEDGE_RBF, 2 * cV_H + 1, cS_H, cV_H);
  }
  GVPp enc2flow = fetch_gvp(cENC_S, cENC_V, cS_H, cV_H);
  GVPp encb[3][2][3];
  for (int i = 0; i < 3; ++i) {
    for (int L = 0; L < 3; ++L)  // msg
      encb[i][0][L] = L ? fetch_gvp(cENC_S, cENC_V, cENC_S, cENC_V)
                        : fetch_gvp(2 * cENC_S + cENC_RBF, 2 * cENC_V + 1, cENC_S, cENC_V);
    for (int L = 0; L < 3; ++L)  // upd
      encb[i][1][L] = L ? fetch_gvp(cENC_S, cENC_V, cENC_S, cENC_V)
                        : fetch_gvp(2 * cENC_S, 2 * cENC_V, cENC_S, cENC_V);
  }
  const float* enc_embed_W = nextParam((long)cP_IN * cENC_S);
  const bf16*  enc_embed_Wb = tob16p(enc_embed_W, cP_IN, cENC_S, 32);
  const float* enc_embed_b = nextParam(cENC_S);
  GVPp head = fetch_gvp(cS_H, cV_H, 0, 1);          // leaves: Wh, Wv, bg
  const float* p_enc_W = nextParam((long)(cS_H + 1) * cS_H);
  const bf16*  p_enc_Wb = tob16p(p_enc_W, cS_H + 1, cS_H, 288);
  const float* p_enc_b = nextParam(cS_H);
  const float* p_ln_b  = nextParam(cS_H);
  const float* p_ln_g  = nextParam(cS_H);
  const float* sc_ln_b = nextParam(cS_H);
  const float* sc_ln_g = nextParam(cS_H);
  const float* sc_sca_W = nextParam(cS_H);
  const float* sc_sca_b = nextParam(cS_H);
  GVPp sc_vec = fetch_gvp(0, 1, 0, cV_H);           // leaves: Wh, Wv, bg
  const float* w_enc_W = nextParam((long)(cW_IN + 1) * cS_H);
  const bf16*  w_enc_Wb = tob16p(w_enc_W, cW_IN + 1, cS_H, 32);
  const float* w_enc_b = nextParam(cS_H);
  const float* w_ln_b  = nextParam(cS_H);
  const float* w_ln_g  = nextParam(cS_H);

  // ---- scratch slabs (sized for worst case E=80000, padded widths) ----
  const long EMAX = 80000;
  bf16*  SIN   = (bf16*)alloc((size_t)EMAX * 640 * 2);
  bf16*  SIN2  = (bf16*)alloc((size_t)EMAX * 320 * 2);
  bf16*  VCM   = (bf16*)alloc((size_t)3 * EMAX * 96 * 2);
  bf16*  VCM2  = (bf16*)alloc((size_t)3 * EMAX * 96 * 2);
  float* VH    = (float*)alloc((size_t)3 * EMAX * 72 * 4);
  bf16*  VHB   = (bf16*)alloc((size_t)3 * EMAX * 96 * 2);
  float* SOUT0 = (float*)alloc((size_t)EMAX * 256 * 4);
  float* SOUT1 = (float*)alloc((size_t)EMAX * 256 * 4);
  float* VOUT0 = (float*)alloc((size_t)EMAX * 96 * 4);
  float* VOUT1 = (float*)alloc((size_t)EMAX * 96 * 4);
  float* VU    = (float*)alloc((size_t)3 * EMAX * 32 * 4);
  float* GATE  = (float*)alloc((size_t)EMAX * 32 * 4);
  bf16*  SGATE = (bf16*)alloc((size_t)EMAX * 256 * 2);

  // persistent node / edge buffers
  float* s_enc  = (float*)alloc((size_t)cN_P * cENC_S * 4);
  float* v_enc  = (float*)alloc((size_t)cN_P * cENC_V * 3 * 4);
  float* aggs_p = (float*)alloc((size_t)cN_P * cENC_S * 4);
  float* aggv_p = (float*)alloc((size_t)cN_P * cENC_V * 3 * 4);
  float* s_lat  = (float*)alloc((size_t)cN_P * cS_H * 4);
  float* v_lat  = (float*)alloc((size_t)cN_P * cV_H * 3 * 4);
  float* s_p    = (float*)alloc((size_t)cN_P * cS_H * 4);
  float* s_w    = (float*)alloc((size_t)cN_W * cS_H * 4);
  float* v_w    = (float*)alloc((size_t)cN_W * cV_H * 3 * 4);
  float* aggs_w = (float*)alloc((size_t)cN_W * cS_H * 4);
  float* aggv_w = (float*)alloc((size_t)cN_W * cV_H * 3 * 4);
  float* pp_rbf = (float*)alloc((size_t)cE_PP * cENC_RBF * 4);
  float* pp_unit= (float*)alloc((size_t)cE_PP * 3 * 4);
  int*   pw_nbr = (int*)alloc((size_t)cE_PW * 4);
  int*   pw_src = (int*)alloc((size_t)cE_PW * 4);
  float* pw_rbf = (float*)alloc((size_t)cE_PW * cEDGE_RBF * 4);
  float* pw_unit= (float*)alloc((size_t)cE_PW * 3 * 4);
  int*   ww_nbr = (int*)alloc((size_t)cE_WW * 4);
  int*   ww_src = (int*)alloc((size_t)cE_WW * 4);
  float* ww_rbf = (float*)alloc((size_t)cE_WW * cEDGE_RBF * 4);
  float* ww_unit= (float*)alloc((size_t)cE_WW * 3 * 4);
  bf16*  PXB    = (bf16*)alloc((size_t)cN_P * 32 * 2);

  auto gemm = [&](const bf16* A, int lda, const bf16* B, int ldb, const float* bias,
                  float* C, int ldc, long M, int N, int K, int act) {
    dim3 g((unsigned)((M + 127) / 128), (unsigned)((N + 31) / 32));
    k_gemm<<<g, 256, 0, stream>>>(A, lda, B, ldb, bias, C, ldc, (int)M, N, K, act);
  };

  auto run_gvp = [&](const GVPp& g, long E, bf16* sin, const bf16* vcm,
                     float* s_out, float* v_out) {
    gemm(vcm, g.vip, g.Whb, g.h, nullptr, VH, g.h, 3 * E, g.h, g.vip, 0);   // Vh
    k_vh_post<<<ew(E * g.hp), 256, 0, stream>>>(VH, E, g.h, g.hp, sin, g.ldw, g.si, VHB);
    gemm(sin, g.ldw, g.Wsb, g.so, g.bs, s_out, g.so, E, g.so, g.ldw, 0);    // s linear
    gemm(VHB, g.hp, g.Wvb, g.vo, nullptr, VU, g.vo, 3 * E, g.vo, g.hp, 0);  // Vu
    k_sigmoid_bf16<<<ew(E * g.so), 256, 0, stream>>>(s_out, SGATE, E * (long)g.so);
    gemm(SGATE, g.so, g.Wgb, g.vo, g.bg, GATE, g.vo, E, g.vo, g.so, 0);     // gate
    k_gate_apply<<<ew(E * g.vo), 256, 0, stream>>>(VU, GATE, E, g.vo, v_out);
    k_relu<<<ew(E * g.so), 256, 0, stream>>>(s_out, E * (long)g.so);        // scalar_act
  };

  auto prep_layer_s = [&](const GVPp& g, long E, const float* s_prev, int sld, bf16* sin) {
    k_copy_f32_bf16_mat<<<ew(E * g.si), 256, 0, stream>>>(s_prev, sld, sin, g.ldw, 0, E, g.si);
    if (g.ldw > g.si + g.h)
      k_zero_cols_bf16<<<ew(E * (long)(g.ldw - g.si - g.h)), 256, 0, stream>>>(
          sin, g.ldw, g.si + g.h, g.ldw, E);
  };

  auto run_chain = [&](const GVPp* ch, long E, float** ms, float** mv) {
    run_gvp(ch[0], E, SIN, VCM, SOUT0, VOUT0);
    prep_layer_s(ch[1], E, SOUT0, ch[0].so, SIN2);
    k_v_to_cm<<<ew(3 * E * (long)ch[1].vip), 256, 0, stream>>>(VOUT0, E, ch[1].vi, ch[1].vip, VCM2);
    run_gvp(ch[1], E, SIN2, VCM2, SOUT1, VOUT1);
    prep_layer_s(ch[2], E, SOUT1, ch[1].so, SIN);
    k_v_to_cm<<<ew(3 * E * (long)ch[2].vip), 256, 0, stream>>>(VOUT1, E, ch[2].vi, ch[2].vip, VCM);
    run_gvp(ch[2], E, SIN, VCM, SOUT0, VOUT0);
    *ms = SOUT0; *mv = VOUT0;
  };

  auto run_messages = [&](const GVPp* ch, const float* s_src, const float* v_src,
                          const float* s_dst, const float* v_dst, int S, int V,
                          const int* sidx, const int* didx, long E,
                          const float* rbf, int R, const float* unit,
                          long n_dst, float* agg_s, float* agg_v, bool zero) {
    k_build_msg_s<<<ew(E * (long)ch[0].ldw), 256, 0, stream>>>(
        s_src, s_dst, S, rbf, R, sidx, didx, E, ch[0].si, ch[0].h, ch[0].ldw, SIN);
    k_build_msg_v<<<ew(3 * E * (long)ch[0].vip), 256, 0, stream>>>(
        v_src, v_dst, V, unit, sidx, didx, E, ch[0].vip, VCM);
    float *ms, *mv;
    run_chain(ch, E, &ms, &mv);
    if (zero) {
      hipMemsetAsync(agg_s, 0, (size_t)n_dst * ch[2].so * 4, stream);
      hipMemsetAsync(agg_v, 0, (size_t)n_dst * ch[2].vo * 3 * 4, stream);
    }
    k_scatter_add<<<ew(E * ch[2].so), 256, 0, stream>>>(ms, didx, E, ch[2].so, agg_s);
    k_scatter_add<<<ew(E * ch[2].vo * 3), 256, 0, stream>>>(mv, didx, E, ch[2].vo * 3, agg_v);
  };

  auto run_update = [&](const GVPp* ch, long n, float* s_node, const float* agg_s,
                        float* v_node, const float* agg_v) {
    int S = ch[0].si / 2, V = ch[0].vi / 2;
    k_copy_f32_bf16_mat<<<ew(n * S), 256, 0, stream>>>(s_node, S, SIN, ch[0].ldw, 0, n, S);
    k_copy_f32_bf16_mat<<<ew(n * S), 256, 0, stream>>>(agg_s, S, SIN, ch[0].ldw, S, n, S);
    if (ch[0].ldw > ch[0].si + ch[0].h)
      k_zero_cols_bf16<<<ew(n * (long)(ch[0].ldw - ch[0].si - ch[0].h)), 256, 0, stream>>>(
          SIN, ch[0].ldw, ch[0].si + ch[0].h, ch[0].ldw, n);
    k_concat_v_cm<<<ew(3 * n * (long)ch[0].vip), 256, 0, stream>>>(v_node, V, agg_v, V, n, ch[0].vip, VCM);
    float *ds, *dv;
    run_chain(ch, n, &ds, &dv);
    k_add<<<ew(n * ch[2].so), 256, 0, stream>>>(s_node, ds, n * (long)ch[2].so);
    k_add<<<ew(n * ch[2].vo * 3), 256, 0, stream>>>(v_node, dv, n * (long)ch[2].vo * 3);
  };

  // ---- protein encoder ----
  k_copy_f32_bf16_mat<<<ew((long)cN_P * cP_IN), 256, 0, stream>>>(protein_x, cP_IN, PXB, 32, 0, cN_P, cP_IN);
  k_zero_cols_bf16<<<ew((long)cN_P * 12), 256, 0, stream>>>(PXB, 32, cP_IN, 32, cN_P);
  gemm(PXB, 32, enc_embed_Wb, cENC_S, enc_embed_b, s_enc, cENC_S, cN_P, cENC_S, 32, 0);
  hipMemsetAsync(v_enc, 0, (size_t)cN_P * cENC_V * 3 * 4, stream);
  k_edge_geom<<<ew(cE_PP), 256, 0, stream>>>(pos_p, pos_p, pp_ei, pp_ei + cE_PP, cE_PP, cENC_RBF, pp_rbf, pp_unit);
  for (int i = 0; i < 3; ++i) {
    run_messages(encb[i][0], s_enc, v_enc, s_enc, v_enc, cENC_S, cENC_V,
                 pp_ei, pp_ei + cE_PP, cE_PP, pp_rbf, cENC_RBF, pp_unit,
                 cN_P, aggs_p, aggv_p, true);
    run_update(encb[i][1], cN_P, s_enc, aggs_p, v_enc, aggv_p);
  }

  // ---- enc2flow projection ----
  prep_layer_s(enc2flow, cN_P, s_enc, enc2flow.si, SIN);
  k_v_to_cm<<<ew(3L * cN_P * enc2flow.vip), 256, 0, stream>>>(v_enc, cN_P, enc2flow.vi, enc2flow.vip, VCM);
  run_gvp(enc2flow, cN_P, SIN, VCM, s_lat, v_lat);

  // ---- time conditioning ----
  k_concat_col_bf16<<<ew((long)cN_P * 288), 256, 0, stream>>>(s_lat, 256, t_in, cN_P, 288, SIN);
  gemm(SIN, 288, p_enc_Wb, 256, p_enc_b, SOUT0, 256, cN_P, 256, 288, 2);
  k_layernorm256<<<cN_P, 256, 0, stream>>>(SOUT0, p_ln_g, p_ln_b, s_p, cN_P);
  k_concat_col_bf16<<<ew((long)cN_W * 32), 256, 0, stream>>>(water_x, 16, t_in, cN_W, 32, SIN);
  gemm(SIN, 32, w_enc_Wb, 256, w_enc_b, SOUT0, 256, cN_W, 256, 32, 2);
  k_layernorm256<<<cN_W, 256, 0, stream>>>(SOUT0, w_ln_g, w_ln_b, s_w, cN_W);

  // ---- self conditioning ----
  hipMemsetAsync(v_w, 0, (size_t)cN_W * cV_H * 3 * 4, stream);
  k_self_cond_vec<<<ew((long)cN_W * 32), 256, 0, stream>>>(x1_pred, pos_w, sc_vec.Wh, sc_vec.Wv, sc_vec.bg, cN_W, v_w);
  k_sc_scalar<<<cN_W, 256, 0, stream>>>(x1_pred, pos_w, sc_sca_W, sc_sca_b, sc_ln_g, sc_ln_b, cN_W, s_w);

  // ---- KNN + edge geometry ----
  k_knn<12><<<cN_P, 128, 0, stream>>>(pos_p, pos_w, cN_P, cN_W, 0, pw_nbr);
  k_make_srcidx<<<ew(cE_PW), 256, 0, stream>>>(pw_src, cE_PW, cK_PW);
  k_knn<8><<<cN_W, 128, 0, stream>>>(pos_w, pos_w, cN_W, cN_W, 1, ww_nbr);
  k_make_srcidx<<<ew(cE_WW), 256, 0, stream>>>(ww_src, cE_WW, cK_WW);
  k_edge_geom<<<ew(cE_PW), 256, 0, stream>>>(pos_p, pos_w, pw_src, pw_nbr, cE_PW, cEDGE_RBF, pw_rbf, pw_unit);
  k_edge_geom<<<ew(cE_WW), 256, 0, stream>>>(pos_w, pos_w, ww_src, ww_nbr, cE_WW, cEDGE_RBF, ww_rbf, ww_unit);

  // ---- hetero message passing (water updated only) ----
  for (int b = 0; b < 4; ++b) {
    run_messages(blocks[b][0], s_p, v_lat, s_w, v_w, cS_H, cV_H,
                 pw_src, pw_nbr, cE_PW, pw_rbf, cEDGE_RBF, pw_unit,
                 cN_W, aggs_w, aggv_w, true);
    run_messages(blocks[b][2], s_w, v_w, s_w, v_w, cS_H, cV_H,
                 ww_src, ww_nbr, cE_WW, ww_rbf, cEDGE_RBF, ww_unit,
                 cN_W, aggs_w, aggv_w, false);
    run_update(blocks[b][1], cN_W, s_w, aggs_w, v_w, aggv_w);
  }

  // ---- vector field head ----
  k_head<<<ew((long)cN_W * 3), 256, 0, stream>>>(v_w, head.Wh, head.Wv, head.bg, cN_W, (float*)d_out);
}
